// GATEncoder_13288628814618
// MI455X (gfx1250) — matrix operand order repaired and checked by reading, not compiled, not run on hardware
//
#include <hip/hip_runtime.h>

// ---------------------------------------------------------------------------
// GAT encoder for MI455X (gfx1250).
//  * Dense projections: v_wmma_f32_16x16x32_bf16 (wave32 WMMA, f32 accum),
//    A tile staged to LDS once per block via the Tensor Data Mover
//    (tensor_load_to_lds + s_wait_tensorcnt), with TDM LDS padding to make
//    the ds_load_b128 fragment reads bank-conflict-free.
//  * Edge softmax/aggregation: bandwidth-bound, float atomics resident in
//    the 192MB L2 (working set ~51MB).
// ---------------------------------------------------------------------------

typedef __attribute__((ext_vector_type(16))) __bf16 v16bf;
typedef __attribute__((ext_vector_type(8)))  float  v8f;
typedef unsigned int u32x4 __attribute__((ext_vector_type(4)));
typedef int          i32x4 __attribute__((ext_vector_type(4)));
typedef int          i32x8 __attribute__((ext_vector_type(8)));

struct alignas(16) U128 { unsigned a, b, c, d; };
union ABFrag { v16bf v; U128 q[2]; };

#define NEG_SLOPE 0.2f

// TDM builtin availability (device pass only; host pass falls back).
#if __has_builtin(__builtin_amdgcn_tensor_load_to_lds) && \
    __has_builtin(__builtin_amdgcn_s_wait_tensorcnt)
#define HAVE_TDM 1
#endif

__device__ __forceinline__ unsigned short f32_to_bf16(float f) {
  unsigned u = __float_as_uint(f);
  u += 0x7FFFu + ((u >> 16) & 1u);   // round-to-nearest-even
  return (unsigned short)(u >> 16);
}

// Monotone float <-> uint mapping so atomicMax(u32) implements float max.
__device__ __forceinline__ unsigned fflip(float f) {
  unsigned u = __float_as_uint(f);
  return (u & 0x80000000u) ? ~u : (u | 0x80000000u);
}
__device__ __forceinline__ float funflip(unsigned u) {
  unsigned v = (u & 0x80000000u) ? (u ^ 0x80000000u) : ~u;
  return __uint_as_float(v);
}
#define FLIP_NEG_INF 0x007FFFFFu   // fflip(-inf)

// --------------------------- elementwise helpers ---------------------------

__global__ __launch_bounds__(256) void k_f32_to_bf16(
    const float* __restrict__ in, unsigned short* __restrict__ out, int n) {
  int i = blockIdx.x * blockDim.x + threadIdx.x;
  if (i < n) out[i] = f32_to_bf16(in[i]);
}

// W [Fin, Fout] row-major f32  ->  Wt [Fout, Fin] row-major bf16
__global__ __launch_bounds__(256) void k_weight_to_bf16_t(
    const float* __restrict__ W, unsigned short* __restrict__ Wt,
    int Fin, int Fout) {
  int i = blockIdx.x * blockDim.x + threadIdx.x;
  if (i >= Fin * Fout) return;
  int o = i / Fin, k = i % Fin;
  Wt[o * Fin + k] = f32_to_bf16(W[k * Fout + o]);
}

__global__ __launch_bounds__(256) void k_zero_f32(float* __restrict__ p, int n) {
  int i = blockIdx.x * blockDim.x + threadIdx.x;
  if (i < n) p[i] = 0.f;
}

// ------------------------------- WMMA GEMM ---------------------------------
// Block = one 16-row band; one wave per 16-col tile.  K = 128 (4 k-steps).
// The 16x128 bf16 A tile (4KB) is DMA'd into LDS once per block by the TDM:
//   D# group0: count=1, lds_addr, global_addr, type=2
//   D# group1: data_size=4B, tensor/tile 64x16 dwords, dim0 stride 64,
//              pad_enable, pad_interval=64 dwords, pad_amount=4 dwords
// The 16B-per-row pad gives an LDS row stride of 272B so the ds_load_b128
// A-fragment reads hit distinct banks across lanes 0..15.
//
// Fragment layouts per CDNA5 ISA §7.12.2:
//   A 16x32:  lane<16 -> row=lane,     K = {0..7, 16..23}
//             lane>=16 -> row=lane-16, K = {8..15, 24..31}
//   B 32x16:  lane<16 -> col=lane, K = {0..7, 16..23} ; lane>=16 -> K = {8..15, 24..31}
//   C/D:      vgpr r -> (M = r + 8*(lane>=16), N = lane&15)
#define LDS_ROW 136   // 128 data + 8 pad (ushorts)

__global__ __launch_bounds__(256) void k_wmma_gemm(
    const unsigned short* __restrict__ A,   // [M, K] bf16 row-major
    const unsigned short* __restrict__ Bt,  // [Fout, K] bf16 (B col-major)
    float* __restrict__ Cout,               // [M, Fout] f32
    int K, int Fout) {
  __shared__ unsigned short ldsA[16 * LDS_ROW];

  const int lane = threadIdx.x & 31;
  const int wave = threadIdx.x >> 5;     // one wave per 16-col tile
  const int row0 = blockIdx.x << 4;
  const int col0 = wave << 4;
  const int half = lane >> 4;            // 0: lanes 0-15, 1: lanes 16-31
  const int rm   = lane & 15;

#ifdef HAVE_TDM
  if (threadIdx.x < 32) {                // one wave issues the DMA
    unsigned lds_base = (unsigned)(unsigned long long)(&ldsA[0]); // LDS offset
    unsigned long long ga =
        (unsigned long long)(const void*)(A + (size_t)row0 * K);
    u32x4 g0 = { 1u,                     // count = 1 (valid descriptor)
                 lds_base,
                 (unsigned)ga,
                 (unsigned)(ga >> 32) | 0x80000000u };  // type = 2 ("image")
    // dword0: data_size=2 (4B) | pad_enable | pad_interval=5 (64 dwords)
    //         | pad_amount=3 (4 dwords)
    i32x8 g1 = { 0x07520000,
                 64 << 16,               // tensor_dim0 = 64 dwords
                 16 << 16,               // tensor_dim1 = 16 rows
                 64 << 16,               // tile_dim0   = 64 dwords
                 16,                     // tile_dim1   = 16 rows
                 64,                     // tensor_dim0_stride = 64 dwords
                 0, 0 };
    i32x4 z4 = { 0, 0, 0, 0 };
#if __clang_major__ >= 23
    i32x8 z8 = { 0, 0, 0, 0, 0, 0, 0, 0 };
    __builtin_amdgcn_tensor_load_to_lds(g0, g1, z4, z4, z8, 0);
#else
    __builtin_amdgcn_tensor_load_to_lds(g0, g1, z4, z4, 0);
#endif
    __builtin_amdgcn_s_wait_tensorcnt((short)0);
  }
#else
  // Fallback: cooperative 16B-chunk staging (same padded layout).
  for (int t = threadIdx.x; t < 16 * 16; t += blockDim.x) {
    int r = t >> 4, cc = t & 15;
    *(U128*)&ldsA[r * LDS_ROW + cc * 8] =
        *(const U128*)(A + (size_t)(row0 + r) * K + cc * 8);
  }
#endif
  __syncthreads();

  const unsigned short* arow = &ldsA[rm * LDS_ROW + half * 8];
  const unsigned short* brow = Bt + (size_t)(col0 + rm) * K + half * 8;

  v8f c = {};
  ABFrag a, b;
#pragma unroll
  for (int k0 = 0; k0 < 128; k0 += 32) {
    a.q[0] = *(const U128*)(arow + k0);        // K = k0+kb .. +7
    a.q[1] = *(const U128*)(arow + k0 + 16);   // K = k0+kb+16 .. +23
    b.q[0] = *(const U128*)(brow + k0);        // K = k0+kb .. +7
    b.q[1] = *(const U128*)(brow + k0 + 16);   // K = k0+kb+16 .. +23
    c = __builtin_amdgcn_wmma_f32_16x16x32_bf16(
        /*neg_a=*/false, a.v, /*neg_b=*/false, b.v,
        /*c_mod=*/(short)0, c, /*reuse_a=*/false, /*reuse_b=*/false);
  }

  float* crow = Cout + (size_t)(row0 + half * 8) * Fout + col0 + rm;
#pragma unroll
  for (int r = 0; r < 8; ++r) crow[(size_t)r * Fout] = c[r];
}

// ------------------------- attention preparation ---------------------------
__global__ __launch_bounds__(256) void k_node_prep(
    const float* __restrict__ h, const float* __restrict__ asrc,
    const float* __restrict__ adst, float* __restrict__ aS,
    float* __restrict__ aD, unsigned* __restrict__ mmax,
    float* __restrict__ denom, int N, int H, int C) {
  int id = blockIdx.x * blockDim.x + threadIdx.x;
  if (id >= N * H) return;
  int n = id / H, hd = id % H;
  const float* hp = h + (size_t)n * H * C + hd * C;
  const float* as = asrc + hd * C;
  const float* ad = adst + hd * C;
  float s = 0.f, d = 0.f;
  for (int c = 0; c < C; ++c) { float v = hp[c]; s += v * as[c]; d += v * ad[c]; }
  aS[id] = s;
  aD[id] = d;
  mmax[id] = FLIP_NEG_INF;
  denom[id] = 0.f;
}

// Pass 1 of segment softmax: per-(dst,head) running max via atomicMax(u32).
__global__ __launch_bounds__(256) void k_edge_max(
    const int* __restrict__ ei, const float* __restrict__ aS,
    const float* __restrict__ aD, unsigned* __restrict__ mmax,
    int E, int N, int H) {
  long long id = (long long)blockIdx.x * blockDim.x + threadIdx.x;
  long long tot = (long long)(E + N) * H;
  if (id >= tot) return;
  int e = (int)(id / H), hd = (int)(id % H);
  int src, dst;
  if (e < E) { src = ei[e]; dst = ei[E + e]; } else { src = dst = e - E; }
  float v = aS[src * H + hd] + aD[dst * H + hd];
  v = (v > 0.f) ? v : NEG_SLOPE * v;
  atomicMax(&mmax[dst * H + hd], fflip(v));
}

// Pass 2: one wave per edge.  Lanes 0..H-1 compute ex=exp(e-m) per head and
// accumulate the softmax denominator; all 32 lanes then do a coalesced gather
// of h[src] and global_atomic_add_f32 scatter into acc[dst].
__global__ __launch_bounds__(256) void k_edge_agg(
    const int* __restrict__ ei, const float* __restrict__ aS,
    const float* __restrict__ aD, const unsigned* __restrict__ mmax,
    const float* __restrict__ hfeat, float* __restrict__ denom,
    float* __restrict__ acc, int E, int N, int H, int C, int Fout) {
  const int wavesPerBlock = blockDim.x >> 5;
  int e = blockIdx.x * wavesPerBlock + (threadIdx.x >> 5);
  if (e >= E + N) return;
  int lane = threadIdx.x & 31;
  int src, dst;
  if (e < E) { src = ei[e]; dst = ei[E + e]; } else { src = dst = e - E; }

  float ex = 0.f;
  if (lane < H) {
    float v = aS[src * H + lane] + aD[dst * H + lane];
    v = (v > 0.f) ? v : NEG_SLOPE * v;
    float m = funflip(mmax[dst * H + lane]);
    ex = __expf(v - m);
    atomicAdd(&denom[dst * H + lane], ex);
  }
  const float* hs = hfeat + (size_t)src * Fout;
  float* ao = acc + (size_t)dst * Fout;
  for (int k = lane; k < Fout; k += 32) {
    float exh = __shfl(ex, k / C, 32);
    atomicAdd(&ao[k], hs[k] * exh);
  }
}

// Finalize: divide by softmax denominator, add bias; ReLU + bf16 for hidden
// layers (feeds next layer's WMMA A operand), f32 for the output layer.
__global__ __launch_bounds__(256) void k_finalize_bf16(
    const float* __restrict__ acc, const float* __restrict__ denom,
    const float* __restrict__ bias, unsigned short* __restrict__ out,
    int N, int H, int C, int Fout) {
  int i = blockIdx.x * blockDim.x + threadIdx.x;
  if (i >= N * Fout) return;
  int n = i / Fout, k = i % Fout;
  float v = acc[i] / (denom[n * H + k / C] + 1e-16f) + bias[k];
  v = (v > 0.f) ? v : 0.f;                 // ReLU
  out[i] = f32_to_bf16(v);
}

__global__ __launch_bounds__(256) void k_finalize_f32(
    const float* __restrict__ acc, const float* __restrict__ denom,
    const float* __restrict__ bias, float* __restrict__ out,
    int N, int H, int C, int Fout) {
  int i = blockIdx.x * blockDim.x + threadIdx.x;
  if (i >= N * Fout) return;
  int n = i / Fout, k = i % Fout;
  out[i] = acc[i] / (denom[n * H + k / C] + 1e-16f) + bias[k];
}

// ------------------------------- launcher ----------------------------------

extern "C" void kernel_launch(void* const* d_in, const int* in_sizes, int n_in,
                              void* d_out, int out_size, void* d_ws, size_t ws_size,
                              hipStream_t stream) {
  (void)n_in; (void)out_size; (void)ws_size;
  const float* x  = (const float*)d_in[0];
  const int*   ei = (const int*)d_in[1];   // [2,E] int32 (JAX x64 disabled)
  const float* W [3] = {(const float*)d_in[2], (const float*)d_in[6],  (const float*)d_in[10]};
  const float* AS[3] = {(const float*)d_in[3], (const float*)d_in[7],  (const float*)d_in[11]};
  const float* AD[3] = {(const float*)d_in[4], (const float*)d_in[8],  (const float*)d_in[12]};
  const float* Bb[3] = {(const float*)d_in[5], (const float*)d_in[9],  (const float*)d_in[13]};

  const int FIN = 128;
  const int N = in_sizes[0] / FIN;         // 50000 (divisible by 16)
  const int E = in_sizes[1] / 2;           // 800000
  const int Hh[3]    = {4, 4, 1};
  const int Cc[3]    = {32, 32, 64};
  const int FoutL[3] = {128, 128, 64};

  // Workspace carve-out (256B aligned): ~67 MB total.
  char* p = (char*)d_ws;
  auto carve = [&](size_t bytes) -> void* {
    void* r = (void*)p; p += (bytes + 255) & ~(size_t)255; return r;
  };
  unsigned short* xb  = (unsigned short*)carve((size_t)N * 128 * 2); // bf16 feats
  unsigned short* wtb = (unsigned short*)carve(128 * 128 * 2);       // bf16 W^T
  float*    hbuf  = (float*)carve((size_t)N * 128 * 4);              // GEMM out
  float*    acc   = (float*)carve((size_t)N * 128 * 4);              // aggregation
  float*    aSrc  = (float*)carve((size_t)N * 4 * 4);
  float*    aDst  = (float*)carve((size_t)N * 4 * 4);
  unsigned* mmax  = (unsigned*)carve((size_t)N * 4 * 4);
  float*    denom = (float*)carve((size_t)N * 4 * 4);

  // Initial feature conversion f32 -> bf16.
  {
    int n = N * FIN;
    k_f32_to_bf16<<<(n + 255) / 256, 256, 0, stream>>>(x, xb, n);
  }

  for (int L = 0; L < 3; ++L) {
    const int H = Hh[L], C = Cc[L], Fout = FoutL[L], Fin = 128;

    { // weight convert + transpose (tiny)
      int n = Fin * Fout;
      k_weight_to_bf16_t<<<(n + 255) / 256, 256, 0, stream>>>(W[L], wtb, Fin, Fout);
    }
    // Dense projection: h = feat @ W via WMMA bf16 (f32 accumulate).
    k_wmma_gemm<<<N / 16, 32 * (Fout / 16), 0, stream>>>(xb, wtb, hbuf, Fin, Fout);

    { // per-(node,head) attention logits + softmax-state init
      int n = N * H;
      k_node_prep<<<(n + 255) / 256, 256, 0, stream>>>(
          hbuf, AS[L], AD[L], aSrc, aDst, mmax, denom, N, H, C);
    }
    { // zero the aggregation accumulator
      int n = N * Fout;
      k_zero_f32<<<(n + 255) / 256, 256, 0, stream>>>(acc, n);
    }
    { // softmax pass 1: per-dst max
      long long n = (long long)(E + N) * H;
      k_edge_max<<<(unsigned)((n + 255) / 256), 256, 0, stream>>>(
          ei, aSrc, aDst, mmax, E, N, H);
    }
    { // softmax pass 2 + weighted aggregation (wave per edge)
      int waves = E + N;
      k_edge_agg<<<(waves + 7) / 8, 256, 0, stream>>>(
          ei, aSrc, aDst, mmax, hbuf, denom, acc, E, N, H, C, Fout);
    }
    // normalize + bias (+ReLU); hidden layers feed bf16 straight to next GEMM
    int n = N * Fout;
    if (L < 2) {
      k_finalize_bf16<<<(n + 255) / 256, 256, 0, stream>>>(
          acc, denom, Bb[L], xb, N, H, C, Fout);
    } else {
      k_finalize_f32<<<(n + 255) / 256, 256, 0, stream>>>(
          acc, denom, Bb[L], (float*)d_out, N, H, C, Fout);
    }
  }
}
